// SCAttention_33105607918052
// MI455X (gfx1250) — matrix operand order repaired and checked by reading, not compiled, not run on hardware
//
#include <hip/hip_runtime.h>
#include <hip/hip_bf16.h>
#include <math.h>

// ---------------------------------------------------------------------------
// SCAttention for MI455X (gfx1250, wave32, WMMA).
// All einsum contractions go through one generic strided batched GEMM
// (C = A*B + bias) using bf16 WMMA 16x16x32 fragments with f32 accumulation
// (v_wmma_f32_16x16x32_bf16). K must be a multiple of 32: the single K=3
// contraction (cosine-feature conv) is zero-padded ahead of time so the hot
// loop is completely branch-free (batched loads, one wait, packed cvt, WMMA).
// K-loop is unrolled x2 (two WMMAs in flight); epilogue is branch-free with
// vectorized bias loads (template HAS_BIAS).
// ---------------------------------------------------------------------------

typedef __attribute__((ext_vector_type(16))) __bf16 v16bf;
typedef __attribute__((ext_vector_type(8)))  float  v8f;

#define BB   2
#define CC   128
#define NN   2048
#define KNBR 9
#define HH   4
#define DH   32

// ------------------------------- GEMM --------------------------------------
// One wave computes one 16x16 C tile. a(m,k)=A[m*sam+k*sak],
// b(k,n)=B[k*sbk+n*sbn], c(m,n)=C[m*scm+n*scn]. Kdim % 32 == 0 required.
// AK: sak==1 (A k-contiguous per lane -> float2 loads)
// BK: sbk==1 (B k-contiguous per lane -> float2 loads)

template <bool AK>
__device__ inline v16bf load_a_frag(const float* __restrict__ A, int mBase,
                                    int k0, long sam, long sak, int lane) {
  v16bf a;
  const int m  = mBase + (lane & 15);
  const int kb = (lane < 16) ? 0 : 8;
#pragma unroll
  for (int p = 0; p < 8; ++p) {
    int kk0 = kb + ((p < 4) ? (2 * p) : (16 + 2 * (p - 4)));
    if (AK) {
      float2 f = *(const float2*)(A + (long)m * sam + (long)(k0 + kk0));
      a[2 * p]     = (__bf16)f.x;
      a[2 * p + 1] = (__bf16)f.y;
    } else {
      float f0 = A[(long)m * sam + (long)(k0 + kk0) * sak];
      float f1 = A[(long)m * sam + (long)(k0 + kk0 + 1) * sak];
      a[2 * p]     = (__bf16)f0;
      a[2 * p + 1] = (__bf16)f1;
    }
  }
  return a;
}

template <bool BK>
__device__ inline v16bf load_b_frag(const float* __restrict__ Bm, int nBase,
                                    int k0, long sbk, long sbn, int lane) {
  v16bf b;
  const int n  = nBase + (lane & 15);
  const int kb = (lane < 16) ? 0 : 8;
#pragma unroll
  for (int p = 0; p < 8; ++p) {
    int kk0 = kb + ((p < 4) ? (2 * p) : (16 + 2 * (p - 4)));
    if (BK) {
      float2 f = *(const float2*)(Bm + (long)n * sbn + (long)(k0 + kk0));
      b[2 * p]     = (__bf16)f.x;
      b[2 * p + 1] = (__bf16)f.y;
    } else {
      float f0 = Bm[(long)(k0 + kk0) * sbk + (long)n * sbn];
      float f1 = Bm[(long)(k0 + kk0 + 1) * sbk + (long)n * sbn];
      b[2 * p]     = (__bf16)f0;
      b[2 * p + 1] = (__bf16)f1;
    }
  }
  return b;
}

template <bool AK, bool BK, bool HB>
__global__ __launch_bounds__(128)
void gemm_bf16_wmma(const float* __restrict__ A, const float* __restrict__ Bm,
                    const float* __restrict__ bias, float* __restrict__ Cm,
                    int M, int Ncols, int Kdim,
                    long sam, long sak, long sbk, long sbn, long scm, long scn,
                    long abat, long bbat, long cbat) {
  const int wid  = threadIdx.x >> 5;
  const int lane = threadIdx.x & 31;
  const long zb  = blockIdx.z;
  A  += zb * abat;
  Bm += zb * bbat;
  Cm += zb * cbat;

  const int mBase = blockIdx.y * 16;
  const int nBase = (blockIdx.x * 4 + wid) * 16;
  if (mBase >= M || nBase >= Ncols) return;   // wave-uniform

  v8f acc;
#pragma unroll
  for (int j = 0; j < 8; ++j) acc[j] = 0.f;

  const int nPf = nBase + (lane & 15);
  int k0 = 0;
  // main loop: 2 independent WMMAs per iteration
  for (; k0 + 64 <= Kdim; k0 += 64) {
    if (k0 + 64 < Kdim)  // pull the K-panel after next into cache
      __builtin_prefetch(&Bm[(long)(k0 + 64) * sbk + (long)nPf * sbn], 0, 3);
    v16bf a0 = load_a_frag<AK>(A, mBase, k0, sam, sak, lane);
    v16bf b0 = load_b_frag<BK>(Bm, nBase, k0, sbk, sbn, lane);
    v16bf a1 = load_a_frag<AK>(A, mBase, k0 + 32, sam, sak, lane);
    v16bf b1 = load_b_frag<BK>(Bm, nBase, k0 + 32, sbk, sbn, lane);
    acc = __builtin_amdgcn_wmma_f32_16x16x32_bf16(
        false, a0, false, b0, (short)0, acc, false, false);
    acc = __builtin_amdgcn_wmma_f32_16x16x32_bf16(
        false, a1, false, b1, (short)0, acc, false, false);
  }
  // remainder (e.g. K==32 attention-score GEMM)
  for (; k0 < Kdim; k0 += 32) {
    v16bf a = load_a_frag<AK>(A, mBase, k0, sam, sak, lane);
    v16bf b = load_b_frag<BK>(Bm, nBase, k0, sbk, sbn, lane);
    acc = __builtin_amdgcn_wmma_f32_16x16x32_bf16(
        false, a, false, b, (short)0, acc, false, false);
  }

  // branch-free epilogue; lane's 8 bias values are contiguous
  float bv[8];
  if (HB) {
    const float* bp = bias + mBase + ((lane < 16) ? 0 : 8);
    float4 b0 = *(const float4*)(bp);
    float4 b1 = *(const float4*)(bp + 4);
    bv[0] = b0.x; bv[1] = b0.y; bv[2] = b0.z; bv[3] = b0.w;
    bv[4] = b1.x; bv[5] = b1.y; bv[6] = b1.z; bv[7] = b1.w;
  }
  const int n = nBase + (lane & 15);
#pragma unroll
  for (int j = 0; j < 8; ++j) {
    int m = mBase + ((lane < 16) ? j : j + 8);
    float v = acc[j];
    if (HB) v += bv[j];
    Cm[(long)m * scm + (long)n * scn] = v;
  }
}

// ------------------------------ KNN ----------------------------------------
__global__ __launch_bounds__(256)
void knn_kernel(const float* __restrict__ coords, int* __restrict__ idx) {
  __shared__ float sx[NN];
  __shared__ float sy[NN];
  const int b = blockIdx.y;
  const int tid = threadIdx.x;
  for (int i = tid; i < NN; i += 256) {
    sx[i] = coords[(long)b * 2 * NN + i];
    sy[i] = coords[(long)b * 2 * NN + NN + i];
  }
  __syncthreads();
  const int n = blockIdx.x * 256 + tid;
  const float px = sx[n], py = sy[n];
  float bd[KNBR + 1];
  int   bi[KNBR + 1];
#pragma unroll
  for (int j = 0; j <= KNBR; ++j) { bd[j] = 3.4e38f; bi[j] = 0; }
  for (int m = 0; m < NN; ++m) {
    float dx = px - sx[m], dy = py - sy[m];
    float d = fmaxf(dx * dx + dy * dy, 1e-12f);
    if (d < bd[KNBR]) {
      int j = KNBR;
      while (j > 0 && d < bd[j - 1]) { bd[j] = bd[j - 1]; bi[j] = bi[j - 1]; --j; }
      bd[j] = d; bi[j] = m;
    }
  }
  long base = ((long)b * NN + n) * KNBR;
#pragma unroll
  for (int j = 0; j < KNBR; ++j) idx[base + j] = bi[j + 1];
}

// -------------------- graph feature: [B,C,N]->[B,2C,N,k] -------------------
__global__ __launch_bounds__(256)
void graph_feature_kernel(const float* __restrict__ feats,
                          const int* __restrict__ idx,
                          float* __restrict__ out, int Cch, long total) {
  long t = (long)blockIdx.x * 256 + threadIdx.x;
  if (t >= total) return;                     // t over B*Cch*N
  int n = (int)(t % NN);
  long bc = t / NN;
  int c = (int)(bc % Cch);
  int b = (int)(bc / Cch);
  float f = feats[t];
  long obase0 = (((long)b * 2 * Cch + c) * NN + n) * KNBR;
  long obase1 = (((long)b * 2 * Cch + Cch + c) * NN + n) * KNBR;
  long ib = ((long)b * NN + n) * KNBR;
#pragma unroll
  for (int j = 0; j < KNBR; ++j) {
    int nb = idx[ib + j];
    float fn = feats[((long)b * Cch + c) * NN + nb];
    out[obase0 + j] = f;
    out[obase1 + j] = fn - f;
  }
}

// ----------------- cosine angle feature: [B,1,N,k] -------------------------
__global__ __launch_bounds__(256)
void cos_kernel(const float* __restrict__ coords, const int* __restrict__ idx,
                float* __restrict__ out, long total) {
  long t = (long)blockIdx.x * 256 + threadIdx.x;
  if (t >= total) return;                     // t over B*N*KNBR
  long bn = t / KNBR;
  int n = (int)(bn % NN);
  int b = (int)(bn / NN);
  int nb = idx[t];
  float ax = coords[(long)b * 2 * NN + n];
  float ay = coords[(long)b * 2 * NN + NN + n];
  float bx = coords[(long)b * 2 * NN + nb];
  float by = coords[(long)b * 2 * NN + NN + nb];
  float dot = ax * bx + ay * by;
  out[t] = dot / (sqrtf(ax * ax + ay * ay) * sqrtf(bx * bx + by * by));
}

// ----------------- zero-padded weight copy: [M,Ksrc]->[M,Kpad] -------------
__global__ __launch_bounds__(256)
void pad_weight_kernel(const float* __restrict__ src, float* __restrict__ dst,
                       int Mrows, int Ksrc, int Kpad) {
  long t = (long)blockIdx.x * 256 + threadIdx.x;
  if (t >= (long)Mrows * Kpad) return;
  int k = (int)(t % Kpad);
  int m = (int)(t / Kpad);
  dst[t] = (k < Ksrc) ? src[(long)m * Ksrc + k] : 0.f;
}

// -------- annu window packs: (1,3)/stride(1,3) conv as GEMM ----------------
// pk[b, kk, n*3+w] = x4[b, kk/3, n, w*3 + kk%3], zero for kk >= Ci*3.
__global__ __launch_bounds__(256)
void pack_annu1(const float* __restrict__ x4, float* __restrict__ pk,
                int Ci, int K1pad, long total) {
  long t = (long)blockIdx.x * 256 + threadIdx.x;
  if (t >= total) return;
  const long N3 = (long)NN * 3;
  long r = t % ((long)K1pad * N3);
  int b = (int)(t / ((long)K1pad * N3));
  int kk = (int)(r / N3);
  int col = (int)(r % N3);
  if (kk >= Ci * 3) { pk[t] = 0.f; return; }
  int c = kk / 3, tt = kk % 3;
  int n = col / 3, w = col % 3;
  pk[t] = x4[(((long)b * Ci + c) * NN + n) * KNBR + w * 3 + tt];
}
// pk2[b, c*3+t, n] = h[b, c, n*3+t]    (h is [B,Ch,N*3])
__global__ __launch_bounds__(256)
void pack_annu2(const float* __restrict__ h, float* __restrict__ pk,
                int Ch, long total) {
  long t = (long)blockIdx.x * 256 + threadIdx.x;
  if (t >= total) return;
  const long K2 = (long)Ch * 3;
  long r = t % (K2 * NN);
  int b = (int)(t / (K2 * NN));
  int kk = (int)(r / NN);
  int n  = (int)(r % NN);
  int c = kk / 3, tt = kk % 3;
  pk[t] = h[((long)b * Ch + c) * (NN * 3) + (long)n * 3 + tt];
}

// ----------------- instance norm + activation (in place) -------------------
// grid = rows (B*Ch); act: 0=none 1=relu 2=leaky-relu(0.2)
__global__ __launch_bounds__(256)
void inorm_act(float* __restrict__ x, long L, int act) {
  __shared__ float s1[256];
  __shared__ float s2[256];
  const int tid = threadIdx.x;
  const long base = (long)blockIdx.x * L;
  float a = 0.f, b = 0.f;
  for (long i = tid; i < L; i += 256) {
    float v = x[base + i];
    a += v; b += v * v;
  }
  s1[tid] = a; s2[tid] = b;
  __syncthreads();
  for (int s = 128; s > 0; s >>= 1) {
    if (tid < s) { s1[tid] += s1[tid + s]; s2[tid] += s2[tid + s]; }
    __syncthreads();
  }
  float mean = s1[0] / (float)L;
  float var  = fmaxf(s2[0] / (float)L - mean * mean, 0.f);
  float inv  = rsqrtf(var + 1e-5f);
  for (long i = tid; i < L; i += 256) {
    float v = (x[base + i] - mean) * inv;
    if (act == 1)      v = fmaxf(v, 0.f);
    else if (act == 2) v = (v > 0.f) ? v : 0.2f * v;
    x[base + i] = v;
  }
}

// --------------------------- max over k ------------------------------------
__global__ __launch_bounds__(256)
void maxk_kernel(const float* __restrict__ in, float* __restrict__ out,
                 long total) {
  long t = (long)blockIdx.x * 256 + threadIdx.x;
  if (t >= total) return;
  float m = in[t * KNBR];
#pragma unroll
  for (int j = 1; j < KNBR; ++j) m = fmaxf(m, in[t * KNBR + j]);
  out[t] = m;
}

// -------------- channel-block copy (+optional add): concat -----------------
__global__ __launch_bounds__(256)
void rows_add_kernel(float* __restrict__ dst, const float* __restrict__ s1,
                     const float* __restrict__ s2, int Cdst, int co, int Cs,
                     long total) {
  long t = (long)blockIdx.x * 256 + threadIdx.x;
  if (t >= total) return;                     // t over B*Cs*N
  int n = (int)(t % NN);
  long bc = t / NN;
  int c = (int)(bc % Cs);
  int b = (int)(bc / Cs);
  float v = s1[t];
  if (s2) v += s2[t];
  dst[((long)b * Cdst + co + c) * NN + n] = v;
}

__global__ __launch_bounds__(256)
void add_ew_kernel(float* __restrict__ out, const float* __restrict__ a,
                   const float* __restrict__ b, long total) {
  long t = (long)blockIdx.x * 256 + threadIdx.x;
  if (t >= total) return;
  out[t] = a[t] + b[t];
}

// ----------------------------- softmax -------------------------------------
__global__ __launch_bounds__(256)
void softmax_rows(float* __restrict__ S, int Ncols, float scale) {
  __shared__ float red[256];
  const int tid = threadIdx.x;
  float* row = S + (long)blockIdx.x * Ncols;
  float mx = -3.4e38f;
  for (int i = tid; i < Ncols; i += 256) mx = fmaxf(mx, row[i] * scale);
  red[tid] = mx; __syncthreads();
  for (int s = 128; s > 0; s >>= 1) {
    if (tid < s) red[tid] = fmaxf(red[tid], red[tid + s]);
    __syncthreads();
  }
  mx = red[0]; __syncthreads();
  float sum = 0.f;
  for (int i = tid; i < Ncols; i += 256) {
    float e = __expf(row[i] * scale - mx);
    row[i] = e; sum += e;
  }
  red[tid] = sum; __syncthreads();
  for (int s = 128; s > 0; s >>= 1) {
    if (tid < s) red[tid] += red[tid + s];
    __syncthreads();
  }
  float inv = 1.f / red[0];
  for (int i = tid; i < Ncols; i += 256) row[i] *= inv;
}

// ===========================================================================
// Host-side orchestration
// ===========================================================================
namespace {

inline long ceil_div(long a, long b) { return (a + b - 1) / b; }

template <bool AK, bool BK>
inline void launch_gemm_b(hipStream_t st, const float* A, const float* Bm,
                          const float* bias, float* Cm, int M, int Nc, int Kd,
                          long sam, long sak, long sbk, long sbn, long scm,
                          long scn, long ab, long bb, long cb, int batch) {
  dim3 grid((unsigned)ceil_div(Nc / 16, 4), (unsigned)(M / 16), (unsigned)batch);
  dim3 blk(128);
  if (bias)
    gemm_bf16_wmma<AK, BK, true><<<grid, blk, 0, st>>>(A, Bm, bias, Cm, M, Nc,
        Kd, sam, sak, sbk, sbn, scm, scn, ab, bb, cb);
  else
    gemm_bf16_wmma<AK, BK, false><<<grid, blk, 0, st>>>(A, Bm, bias, Cm, M, Nc,
        Kd, sam, sak, sbk, sbn, scm, scn, ab, bb, cb);
}

inline void launch_gemm(hipStream_t st, const float* A, const float* Bm,
                        const float* bias, float* Cm, int M, int Nc, int Kd,
                        long sam, long sak, long sbk, long sbn, long scm,
                        long scn, long ab, long bb, long cb, int batch) {
  if (sak == 1) {
    if (sbk == 1)
      launch_gemm_b<true, true>(st, A, Bm, bias, Cm, M, Nc, Kd, sam, sak, sbk,
                                sbn, scm, scn, ab, bb, cb, batch);
    else
      launch_gemm_b<true, false>(st, A, Bm, bias, Cm, M, Nc, Kd, sam, sak, sbk,
                                 sbn, scm, scn, ab, bb, cb, batch);
  } else {
    if (sbk == 1)
      launch_gemm_b<false, true>(st, A, Bm, bias, Cm, M, Nc, Kd, sam, sak, sbk,
                                 sbn, scm, scn, ab, bb, cb, batch);
    else
      launch_gemm_b<false, false>(st, A, Bm, bias, Cm, M, Nc, Kd, sam, sak,
                                  sbk, sbn, scm, scn, ab, bb, cb, batch);
  }
}

inline void launch_ew(hipStream_t st, long total, float* out, const float* a,
                      const float* b) {
  add_ew_kernel<<<(unsigned)ceil_div(total, 256), 256, 0, st>>>(out, a, b, total);
}

struct Ws {
  int*   idx1;
  float *g4a, *g4b, *g4c;      // [B,256,N,9] each; g4c doubles as annu pack buf
  float *h1;                   // [B,128,N*3]
  float *cosb;                 // [B,1,N,9]
  float *wpad;                 // [128,32] zero-padded gw1
  float *x1buf, *fang, *x2buf, *x3buf, *x1o, *tmpCN;
  float *x2o;                  // [B,256,N]
  float *cat3;                 // [B,384,N]
  float *cat4;                 // [B,512,N]
  float *sa0, *sa1;
  float *qb, *kb, *vb, *attnx, *msgb;
  float *cat2, *mh;            // [B,256,N]
  float *scores;               // [N,N]
};

inline Ws carve_ws(void* d_ws) {
  char* p = (char*)d_ws;
  size_t o = 0;
  auto take = [&](size_t bytes) {
    char* r = p + o;
    o = (o + bytes + 255) & ~((size_t)255);
    return r;
  };
  const size_t CN  = (size_t)BB * CC * NN * 4;             // 2 MB
  const size_t BIG = (size_t)BB * 2 * CC * NN * KNBR * 4;  // 37.7 MB
  Ws w;
  w.idx1  = (int*)take((size_t)BB * NN * KNBR * 4);
  w.g4a   = (float*)take(BIG);
  w.g4b   = (float*)take(BIG);
  w.g4c   = (float*)take(BIG);   // >= B*768*(N*3)*4, also conv scratch
  w.h1    = (float*)take((size_t)BB * CC * NN * 3 * 4);
  w.cosb  = (float*)take((size_t)BB * NN * KNBR * 4);
  w.wpad  = (float*)take((size_t)CC * 32 * 4);
  w.x1buf = (float*)take(CN);
  w.fang  = (float*)take(CN);
  w.x2buf = (float*)take(CN);
  w.x3buf = (float*)take(CN);
  w.x1o   = (float*)take(CN);
  w.tmpCN = (float*)take(CN);
  w.x2o   = (float*)take(2 * CN);
  w.cat3  = (float*)take(3 * CN);
  w.cat4  = (float*)take(4 * CN);
  w.sa0   = (float*)take(CN);
  w.sa1   = (float*)take(CN);
  w.qb    = (float*)take(CN);
  w.kb    = (float*)take(CN);
  w.vb    = (float*)take(CN);
  w.attnx = (float*)take(CN);
  w.msgb  = (float*)take(CN);
  w.cat2  = (float*)take(2 * CN);
  w.mh    = (float*)take(2 * CN);
  w.scores = (float*)take((size_t)NN * NN * 4);
  return w;
}

// annu: x4 [B,Ci,N,9] -> out [B,128,N].
// w1pad is [128, K1pad] row-major (already zero-padded when Ci*3 % 32 != 0).
inline void run_annu(hipStream_t st, const Ws& w, const float* x4, int Ci,
                     const float* w1pad, const float* b1, const float* w2,
                     const float* b2, float* out) {
  const long N3 = (long)NN * 3;
  const int K1pad = (int)((((long)Ci * 3) + 31) / 32 * 32);
  float* pk = w.g4c;
  {
    long total = (long)BB * K1pad * N3;
    pack_annu1<<<(unsigned)ceil_div(total, 256), 256, 0, st>>>(x4, pk, Ci,
                                                               K1pad, total);
  }
  // stage 1: [128 x K1pad] * [K1pad x N*3] + b1
  launch_gemm(st, w1pad, pk, b1, w.h1, CC, (int)N3, K1pad,
              K1pad, 1, N3, 1, N3, 1, 0, (long)K1pad * N3, (long)CC * N3, BB);
  inorm_act<<<BB * CC, 256, 0, st>>>(w.h1, N3, 1);
  // stage 2: [128 x 384] * [384 x N] + b2
  {
    long total = (long)BB * (CC * 3) * NN;
    pack_annu2<<<(unsigned)ceil_div(total, 256), 256, 0, st>>>(w.h1, pk, CC, total);
  }
  launch_gemm(st, w2, pk, b2, out, CC, NN, CC * 3,
              CC * 3, 1, NN, 1, NN, 1, 0, (long)CC * 3 * NN, (long)CC * NN, BB);
  inorm_act<<<BB * CC, 256, 0, st>>>(out, NN, 1);
}

} // namespace

extern "C" void kernel_launch(void* const* d_in, const int* in_sizes, int n_in,
                              void* d_out, int out_size, void* d_ws,
                              size_t ws_size, hipStream_t stream) {
  (void)in_sizes; (void)n_in; (void)out_size; (void)ws_size;
  const float* desc0   = (const float*)d_in[0];
  const float* desc1   = (const float*)d_in[1];
  const float* coords0 = (const float*)d_in[2];
  const float* coords1 = (const float*)d_in[3];
  const float* conv1_w = (const float*)d_in[4];   // [128,256]
  const float* conv2_w = (const float*)d_in[5];   // [256,256]
  const float* conv3_w = (const float*)d_in[6];   // [128,384]
  const float* conv3o_w= (const float*)d_in[7];   // [128,512]
  const float* a1w1 = (const float*)d_in[8];   const float* a1b1 = (const float*)d_in[9];
  const float* a1w2 = (const float*)d_in[10];  const float* a1b2 = (const float*)d_in[11];
  const float* a2w1 = (const float*)d_in[12];  const float* a2b1 = (const float*)d_in[13];
  const float* a2w2 = (const float*)d_in[14];  const float* a2b2 = (const float*)d_in[15];
  const float* gw1  = (const float*)d_in[16];  const float* gb1  = (const float*)d_in[17];
  const float* gw2  = (const float*)d_in[18];  const float* gb2  = (const float*)d_in[19];
  const float* wq = (const float*)d_in[20]; const float* bq = (const float*)d_in[21];
  const float* wk = (const float*)d_in[22]; const float* bk = (const float*)d_in[23];
  const float* wv = (const float*)d_in[24]; const float* bv = (const float*)d_in[25];
  const float* wm = (const float*)d_in[26]; const float* bm = (const float*)d_in[27];
  const float* mw1 = (const float*)d_in[28]; const float* mb1 = (const float*)d_in[29];
  const float* mw2 = (const float*)d_in[30]; const float* mb2 = (const float*)d_in[31];

  Ws w = carve_ws(d_ws);
  const long CNtot = (long)BB * CC * NN;
  const long NK    = (long)NN * KNBR;
  float* out0 = (float*)d_out;
  float* out1 = out0 + CNtot;

  // gw1 is [128,3]: zero-pad once to [128,32] so the GEMM needs no K guards.
  pad_weight_kernel<<<(unsigned)ceil_div((long)CC * 32, 256), 256, 0, stream>>>(
      gw1, w.wpad, CC, 3, 32);

  // ------------------- self attention (shared lambda) ----------------------
  auto self_attn = [&](const float* coords, const float* feats, float* out) {
    // KNN is always on coords -> idx identical for all three graph features.
    knn_kernel<<<dim3(NN / 256, BB), 256, 0, stream>>>(coords, w.idx1);
    // x1_old = graph_feature(feats)
    graph_feature_kernel<<<(unsigned)ceil_div(CNtot, 256), 256, 0, stream>>>(
        feats, w.idx1, w.g4a, CC, CNtot);
    // x1 = annu(x1_old)
    run_annu(stream, w, w.g4a, 2 * CC, a1w1, a1b1, a1w2, a1b2, w.x1buf);
    // f_ang = annu(cos)  (f_ang1 == f_ang2 since idx1 == idx2)
    cos_kernel<<<(unsigned)ceil_div((long)BB * NK, 256), 256, 0, stream>>>(
        coords, w.idx1, w.cosb, (long)BB * NK);
    run_annu(stream, w, w.cosb, 1, w.wpad, gb1, gw2, gb2, w.fang);
    // x2 = annu(graph_feature(x1))
    graph_feature_kernel<<<(unsigned)ceil_div(CNtot, 256), 256, 0, stream>>>(
        w.x1buf, w.idx1, w.g4b, CC, CNtot);
    run_annu(stream, w, w.g4b, 2 * CC, a2w1, a2b1, a2w2, a2b2, w.x2buf);
    // x3 = lrelu(inorm(conv3 * [x0; x1+fang; x2+fang]))
    rows_add_kernel<<<(unsigned)ceil_div(CNtot, 256), 256, 0, stream>>>(
        w.cat3, feats, nullptr, 3 * CC, 0, CC, CNtot);
    rows_add_kernel<<<(unsigned)ceil_div(CNtot, 256), 256, 0, stream>>>(
        w.cat3, w.x1buf, w.fang, 3 * CC, CC, CC, CNtot);
    rows_add_kernel<<<(unsigned)ceil_div(CNtot, 256), 256, 0, stream>>>(
        w.cat3, w.x2buf, w.fang, 3 * CC, 2 * CC, CC, CNtot);
    launch_gemm(stream, conv3_w, w.cat3, nullptr, w.x3buf, CC, NN, 3 * CC,
                3 * CC, 1, NN, 1, NN, 1, 0, (long)3 * CC * NN, (long)CC * NN, BB);
    inorm_act<<<BB * CC, 256, 0, stream>>>(w.x3buf, NN, 2);
    // x1o = max_k(lrelu(inorm(conv1 * x1_old)))
    launch_gemm(stream, conv1_w, w.g4a, nullptr, w.g4c, CC, (int)NK, 2 * CC,
                2 * CC, 1, NK, 1, NK, 1, 0, (long)2 * CC * NK, (long)CC * NK, BB);
    inorm_act<<<BB * CC, 256, 0, stream>>>(w.g4c, NK, 2);
    maxk_kernel<<<(unsigned)ceil_div(CNtot, 256), 256, 0, stream>>>(w.g4c, w.x1o, CNtot);
    // x2o = max_k(lrelu(inorm(conv2 * graph_feature(x1o))))
    graph_feature_kernel<<<(unsigned)ceil_div(CNtot, 256), 256, 0, stream>>>(
        w.x1o, w.idx1, w.g4b, CC, CNtot);
    launch_gemm(stream, conv2_w, w.g4b, nullptr, w.g4c, 2 * CC, (int)NK, 2 * CC,
                2 * CC, 1, NK, 1, NK, 1, 0, (long)2 * CC * NK, (long)2 * CC * NK, BB);
    inorm_act<<<BB * 2 * CC, 256, 0, stream>>>(w.g4c, NK, 2);
    maxk_kernel<<<(unsigned)ceil_div(2 * CNtot, 256), 256, 0, stream>>>(w.g4c, w.x2o, 2 * CNtot);
    // x3o = lrelu(inorm(conv3o * [x0; x1o; x2o]))
    rows_add_kernel<<<(unsigned)ceil_div(CNtot, 256), 256, 0, stream>>>(
        w.cat4, feats, nullptr, 4 * CC, 0, CC, CNtot);
    rows_add_kernel<<<(unsigned)ceil_div(CNtot, 256), 256, 0, stream>>>(
        w.cat4, w.x1o, nullptr, 4 * CC, CC, CC, CNtot);
    rows_add_kernel<<<(unsigned)ceil_div(2 * CNtot, 256), 256, 0, stream>>>(
        w.cat4, w.x2o, nullptr, 4 * CC, 2 * CC, 2 * CC, 2 * CNtot);
    launch_gemm(stream, conv3o_w, w.cat4, nullptr, w.tmpCN, CC, NN, 4 * CC,
                4 * CC, 1, NN, 1, NN, 1, 0, (long)4 * CC * NN, (long)CC * NN, BB);
    inorm_act<<<BB * CC, 256, 0, stream>>>(w.tmpCN, NN, 2);
    launch_ew(stream, CNtot, out, w.x3buf, w.tmpCN);
  };

  // ------------------- cross attention propagation -------------------------
  auto attn_prop = [&](const float* x, const float* src, float* dst) {
    // Q/K/V projections: [128x128] * [128xN] + bias, batched over B
    launch_gemm(stream, wq, x,   bq, w.qb, CC, NN, CC, CC, 1, NN, 1, NN, 1,
                0, (long)CC * NN, (long)CC * NN, BB);
    launch_gemm(stream, wk, src, bk, w.kb, CC, NN, CC, CC, 1, NN, 1, NN, 1,
                0, (long)CC * NN, (long)CC * NN, BB);
    launch_gemm(stream, wv, src, bv, w.vb, CC, NN, CC, CC, 1, NN, 1, NN, 1,
                0, (long)CC * NN, (long)CC * NN, BB);
    const float scale = 1.f / sqrtf((float)DH);
    for (int b = 0; b < BB; ++b) {
      for (int h = 0; h < HH; ++h) {
        const long hb = (long)b * CC * NN + (long)h * NN;  // channel c = d*H+h
        // scores[n,m] = sum_d q[d,n] * k[d,m]
        launch_gemm(stream, w.qb + hb, w.kb + hb, nullptr, w.scores,
                    NN, NN, DH,
                    /*sam*/ 1, /*sak*/ (long)HH * NN,
                    /*sbk*/ (long)HH * NN, /*sbn*/ 1,
                    /*scm*/ NN, /*scn*/ 1, 0, 0, 0, 1);
        softmax_rows<<<NN, 256, 0, stream>>>(w.scores, NN, scale);
        // out[d,n] = sum_m v[d,m] * prob[n,m]
        launch_gemm(stream, w.vb + hb, w.scores, nullptr, w.attnx + hb,
                    DH, NN, NN,
                    /*sam*/ (long)HH * NN, /*sak*/ 1,
                    /*sbk*/ 1, /*sbn*/ NN,
                    /*scm*/ (long)HH * NN, /*scn*/ 1, 0, 0, 0, 1);
      }
    }
    // merge heads + MLP
    launch_gemm(stream, wm, w.attnx, bm, w.msgb, CC, NN, CC, CC, 1, NN, 1, NN,
                1, 0, (long)CC * NN, (long)CC * NN, BB);
    rows_add_kernel<<<(unsigned)ceil_div(CNtot, 256), 256, 0, stream>>>(
        w.cat2, x, nullptr, 2 * CC, 0, CC, CNtot);
    rows_add_kernel<<<(unsigned)ceil_div(CNtot, 256), 256, 0, stream>>>(
        w.cat2, w.msgb, nullptr, 2 * CC, CC, CC, CNtot);
    launch_gemm(stream, mw1, w.cat2, mb1, w.mh, 2 * CC, NN, 2 * CC,
                2 * CC, 1, NN, 1, NN, 1, 0, (long)2 * CC * NN, (long)2 * CC * NN, BB);
    inorm_act<<<BB * 2 * CC, 256, 0, stream>>>(w.mh, NN, 1);
    launch_gemm(stream, mw2, w.mh, mb2, w.tmpCN, CC, NN, 2 * CC,
                2 * CC, 1, NN, 1, NN, 1, 0, (long)2 * CC * NN, (long)CC * NN, BB);
    launch_ew(stream, CNtot, dst, x, w.tmpCN);   // residual
  };

  // --------------------------- full pipeline -------------------------------
  self_attn(coords0, desc0, w.sa0);
  self_attn(coords1, desc1, w.sa1);
  attn_prop(w.sa0, w.sa1, out0);
  attn_prop(w.sa1, out0, out1);
}
